// GPT_90443421319198
// MI455X (gfx1250) — matrix operand order repaired and checked by reading, not compiled, not run on hardware
//
#include <hip/hip_runtime.h>
#include <hip/hip_bf16.h>

// Problem constants (match reference)
#define VOCAB 50257
#define TSEQ  1024
#define CDIM  1024
#define NH    16
#define HD    64
#define BB    2

typedef __bf16 bf16;
typedef __attribute__((ext_vector_type(16))) __bf16 v16bf;
typedef __attribute__((ext_vector_type(8)))  __bf16 v8bf;
typedef __attribute__((ext_vector_type(4)))  __bf16 v4bf;
typedef __attribute__((ext_vector_type(8)))  float  v8f;
typedef __attribute__((ext_vector_type(4)))  float  f32x4;

// ---------------------------------------------------------------------------
// WMMA fragment loaders (wave32, 16x16x32 bf16).
// A (16x32, MxK): lane L = row M=L%16, holds K = hi*16 + {0..15} packed 2/VGPR
//                 (hi = L>=16).
// B (32x16, KxN): lane L = col N=L%16, holds K = hi*16 + {0..15} packed 2/VGPR.
// Both reduce to two contiguous 16-byte loads per lane when the source is
// stored with the fragment's "row" contiguous in K.
// ---------------------------------------------------------------------------
static __device__ inline v16bf load_a_frag(const bf16* __restrict__ base, int ld, int lane) {
  const int r  = lane & 15;
  const int hi = lane >> 4;
  const bf16* p = base + r * ld + hi * 16;
  v8bf lo = *reinterpret_cast<const v8bf*>(p);
  v8bf hh = *reinterpret_cast<const v8bf*>(p + 8);
  v16bf f;
#pragma unroll
  for (int i = 0; i < 8; ++i) { f[i] = lo[i]; f[i + 8] = hh[i]; }
  return f;
}

static __device__ inline v16bf load_b_frag(const bf16* __restrict__ baseT, int ld, int lane) {
  const int n  = lane & 15;
  const int hi = lane >> 4;
  const bf16* p = baseT + n * ld + hi * 16;
  v8bf lo = *reinterpret_cast<const v8bf*>(p);
  v8bf hh = *reinterpret_cast<const v8bf*>(p + 8);
  v16bf f;
#pragma unroll
  for (int i = 0; i < 8; ++i) { f[i] = lo[i]; f[i + 8] = hh[i]; }
  return f;
}

// B-fragment from an LDS panel stored with per-row rotate swizzle:
// element (n, k) lives at n*1024 + ((k + 8n) & 1023). 16B chunks stay aligned,
// and the 16 fragment lanes land on banks 4n apart -> conflict-free.
static __device__ inline v16bf lds_b_frag_sw(const bf16* base, int kk, int lane) {
  const int n  = lane & 15;
  const int hi = lane >> 4;
  const int k0 = kk + hi * 16;
  const bf16* rowp = base + n * 1024;
  const int rot = n * 8;
  v8bf lo = *reinterpret_cast<const v8bf*>(rowp + ((k0 + rot) & 1023));
  v8bf hh = *reinterpret_cast<const v8bf*>(rowp + ((k0 + 8 + rot) & 1023));
  v16bf f;
#pragma unroll
  for (int i = 0; i < 8; ++i) { f[i] = lo[i]; f[i + 8] = hh[i]; }
  return f;
}

static __device__ inline v8f wmma_bf16(v16bf a, v16bf b, v8f c) {
  return __builtin_amdgcn_wmma_f32_16x16x32_bf16(false, a, false, b, (short)0, c, false, false);
}

// ---------------------------------------------------------------------------
// Kernel 1: x = tok_emb[idx] + pos_emb, stored bf16 [B,T,C]
// ---------------------------------------------------------------------------
__global__ void k_embed(const int* __restrict__ idx,
                        const float* __restrict__ tok_emb,
                        const float* __restrict__ pos_emb,
                        bf16* __restrict__ xbf) {
  const int bt = blockIdx.x;           // 0 .. B*T-1
  const int t  = bt & (TSEQ - 1);
  const int c0 = threadIdx.x * 4;
  const int token = idx[bt];
  f32x4 a = *reinterpret_cast<const f32x4*>(tok_emb + (size_t)token * CDIM + c0);
  f32x4 p = *reinterpret_cast<const f32x4*>(pos_emb + (size_t)t * CDIM + c0);
  v4bf o;
#pragma unroll
  for (int i = 0; i < 4; ++i) o[i] = (bf16)(a[i] + p[i]);
  *reinterpret_cast<v4bf*>(xbf + (size_t)bt * CDIM + c0) = o;
}

// ---------------------------------------------------------------------------
// Kernel 2: convert Wq/Wk/Wv [H,C,HD] fp32 -> bf16 transposed [H,HD,C]
// ---------------------------------------------------------------------------
__global__ void k_convw(const float* __restrict__ Wq, const float* __restrict__ Wk,
                        const float* __restrict__ Wv,
                        bf16* __restrict__ wqT, bf16* __restrict__ wkT,
                        bf16* __restrict__ wvT) {
  const int i = blockIdx.x * blockDim.x + threadIdx.x;   // over H*C*HD
  const int h = i / (CDIM * HD);
  const int r = i - h * (CDIM * HD);
  const int c = r / HD;
  const int d = r - c * HD;
  const int o = (h * HD + d) * CDIM + c;
  wqT[o] = (bf16)Wq[i];
  wkT[o] = (bf16)Wk[i];
  wvT[o] = (bf16)Wv[i];
}

// ---------------------------------------------------------------------------
// Kernel 3: QKV projection. Per (b,h,m-tile): 16(M) x 64(N) x 1024(K).
// Block = 128 threads (4 waves, one 16-wide N tile each). Shared A fragment,
// three WMMAs per K step. Stores q,k as [b,h,t,d]; v transposed as [b,h,d,t]
// (the accumulator's 8 rows per lane are contiguous in t -> one 16B store).
// ---------------------------------------------------------------------------
__global__ void k_qkv(const bf16* __restrict__ xbf,
                      const bf16* __restrict__ wqT, const bf16* __restrict__ wkT,
                      const bf16* __restrict__ wvT,
                      bf16* __restrict__ q, bf16* __restrict__ k,
                      bf16* __restrict__ vT) {
  const int b = blockIdx.z, h = blockIdx.y, m0 = blockIdx.x * 16;
  const int lane = threadIdx.x & 31;
  const int wv_  = threadIdx.x >> 5;     // 0..3
  const int d0 = wv_ * 16;
  const int hi = lane >> 4;
  const int col = lane & 15;

  const bf16* xrow = xbf + ((size_t)(b * TSEQ + m0)) * CDIM;
  const size_t wbase = ((size_t)h * HD + d0) * CDIM;

  v8f aq = (v8f)0.f, ak = (v8f)0.f, av = (v8f)0.f;
#pragma unroll 4
  for (int kk = 0; kk < CDIM; kk += 32) {
    v16bf a = load_a_frag(xrow + kk, CDIM, lane);
    aq = wmma_bf16(a, load_b_frag(wqT + wbase + kk, CDIM, lane), aq);
    ak = wmma_bf16(a, load_b_frag(wkT + wbase + kk, CDIM, lane), ak);
    av = wmma_bf16(a, load_b_frag(wvT + wbase + kk, CDIM, lane), av);
  }

  const size_t bh = (size_t)b * NH + h;
  const int d = d0 + col;
  // q, k: [bh, t, d]
#pragma unroll
  for (int g = 0; g < 8; ++g) {
    const int t = m0 + hi * 8 + g;
    q[(bh * TSEQ + t) * HD + d] = (bf16)aq[g];
    k[(bh * TSEQ + t) * HD + d] = (bf16)ak[g];
  }
  // vT: [bh, d, t], rows g are contiguous t -> vector store
  v8bf vo;
#pragma unroll
  for (int g = 0; g < 8; ++g) vo[g] = (bf16)av[g];
  *reinterpret_cast<v8bf*>(vT + (bh * HD + d) * TSEQ + m0 + hi * 8) = vo;
}

// ---------------------------------------------------------------------------
// Kernel 4: causal attention, flash-style, one wave per (b,h,16-row q tile).
// Per 32-wide s-chunk: 4 score WMMAs, online softmax (shfl_xor row reduce,
// v_exp_f32), P -> LDS (layout transpose) -> A fragment, 4 P@V WMMAs.
// Output (o/l) concat-heads bf16 [B,T,C].
// ---------------------------------------------------------------------------
__global__ void k_attn(const bf16* __restrict__ q, const bf16* __restrict__ k,
                       const bf16* __restrict__ vT, bf16* __restrict__ out) {
  __shared__ __align__(16) bf16 pL[16 * 40];   // padded: stride 40 elems (80B)

  const int b = blockIdx.z, h = blockIdx.y, qt = blockIdx.x;
  const int m0 = qt * 16;
  const int lane = threadIdx.x & 31;
  const int hi = lane >> 4;
  const int col = lane & 15;
  const size_t bh = (size_t)b * NH + h;

  const bf16* qbase = q + (bh * TSEQ + m0) * HD;
  const v16bf qa0 = load_a_frag(qbase, HD, lane);
  const v16bf qa1 = load_a_frag(qbase + 32, HD, lane);

  float mrun[8], lrun[8];
  v8f o[4];
#pragma unroll
  for (int g = 0; g < 8; ++g) { mrun[g] = -1e30f; lrun[g] = 0.f; }
#pragma unroll
  for (int j = 0; j < 4; ++j) o[j] = (v8f)0.f;

  const float sc = 0.03125f;               // C^-0.5 = 1024^-0.5
  const int nch = (qt + 2) >> 1;           // ceil((qt+1)/2) 32-wide chunks

  for (int ch = 0; ch < nch; ++ch) {
    const int s0 = ch * 32;
    const bf16* kb0 = k + (bh * TSEQ + s0) * HD;
    const bf16* kb1 = kb0 + 16 * HD;
    v8f sa0 = (v8f)0.f, sa1 = (v8f)0.f;
    sa0 = wmma_bf16(qa0, load_b_frag(kb0, HD, lane), sa0);
    sa0 = wmma_bf16(qa1, load_b_frag(kb0 + 32, HD, lane), sa0);
    sa1 = wmma_bf16(qa0, load_b_frag(kb1, HD, lane), sa1);
    sa1 = wmma_bf16(qa1, load_b_frag(kb1 + 32, HD, lane), sa1);

    float p0[8], p1[8];
#pragma unroll
    for (int g = 0; g < 8; ++g) {
      const int trow = m0 + hi * 8 + g;
      float v0 = (s0 + col      <= trow) ? sa0[g] * sc : -1e30f;
      float v1 = (s0 + 16 + col <= trow) ? sa1[g] * sc : -1e30f;
      float cm = fmaxf(v0, v1);
      cm = fmaxf(cm, __shfl_xor(cm, 1, 32));
      cm = fmaxf(cm, __shfl_xor(cm, 2, 32));
      cm = fmaxf(cm, __shfl_xor(cm, 4, 32));
      cm = fmaxf(cm, __shfl_xor(cm, 8, 32));
      const float mn = fmaxf(mrun[g], cm);
      p0[g] = __expf(v0 - mn);
      p1[g] = __expf(v1 - mn);
      float rs = p0[g] + p1[g];
      rs += __shfl_xor(rs, 1, 32);
      rs += __shfl_xor(rs, 2, 32);
      rs += __shfl_xor(rs, 4, 32);
      rs += __shfl_xor(rs, 8, 32);
      const float sOld = __expf(mrun[g] - mn);
      lrun[g] = lrun[g] * sOld + rs;
      mrun[g] = mn;
#pragma unroll
      for (int j = 0; j < 4; ++j) o[j][g] *= sOld;
    }

    __syncthreads();
#pragma unroll
    for (int g = 0; g < 8; ++g) {
      const int r = hi * 8 + g;
      pL[r * 40 + col]      = (bf16)p0[g];
      pL[r * 40 + 16 + col] = (bf16)p1[g];
    }
    __syncthreads();
    const v16bf pa = load_a_frag(pL, 40, lane);

    const bf16* vtb = vT + bh * HD * TSEQ + s0;
#pragma unroll
    for (int j = 0; j < 4; ++j)
      o[j] = wmma_bf16(pa, load_b_frag(vtb + (size_t)(j * 16) * TSEQ, TSEQ, lane), o[j]);
  }

  // out[b, t, h*HD + d] = o / l
#pragma unroll
  for (int g = 0; g < 8; ++g) {
    const float inv = 1.0f / lrun[g];
    const size_t row = (size_t)(b * TSEQ + m0 + hi * 8 + g) * CDIM + h * HD;
#pragma unroll
    for (int j = 0; j < 4; ++j) out[row + j * 16 + col] = (bf16)(o[j][g] * inv);
  }
}

// ---------------------------------------------------------------------------
// Kernel 5: LM head. logits[2048, V] = attn_out[2048,1024] @ W_lm^T + b.
// One block per 64-column vocab panel: stage the fp32 weight panel once into
// a 128KB swizzled bf16 LDS image (gfx1250 allows up to 320KB LDS/workgroup),
// then 8 waves sweep all M with 2Mx4N register tiling: each A fragment feeds
// 4 WMMAs, each LDS B fragment feeds 2, 8 independent accumulator chains.
// W_lm is read from HBM exactly once; A L2 traffic ~ 786 blocks x 4MB.
// ---------------------------------------------------------------------------
#define NPAN 4   // 16-column panels per block
__global__ void k_lmhead(const bf16* __restrict__ A, const float* __restrict__ W,
                         const float* __restrict__ bias, float* __restrict__ out) {
  __shared__ __align__(16) bf16 bS[NPAN * 16 * 1024];   // 128 KB, swizzled rows

  const int n0  = blockIdx.x * (NPAN * 16);
  const int tid = threadIdx.x;

  // Stage: 64 rows x 1024 cols fp32 -> bf16 swizzled LDS. 256 elems/thread.
  {
    const int row   = tid >> 2;            // 0..63 (panel column index)
    const int cseg  = (tid & 3) * 256;
    const int nt    = row >> 4;
    const int n     = row & 15;
    const int wrow  = min(n0 + row, VOCAB - 1);
    const float* src = W + (size_t)wrow * CDIM + cseg;
    bf16* dst = bS + nt * 16384 + n * 1024;
    const int rot = n * 8;
#pragma unroll 8
    for (int i = 0; i < 32; ++i) {
      f32x4 u = reinterpret_cast<const f32x4*>(src)[i * 2];
      f32x4 w = reinterpret_cast<const f32x4*>(src)[i * 2 + 1];
      v8bf o8;
#pragma unroll
      for (int t = 0; t < 4; ++t) { o8[t] = (bf16)u[t]; o8[t + 4] = (bf16)w[t]; }
      const int klog = cseg + i * 8;
      *reinterpret_cast<v8bf*>(dst + ((klog + rot) & 1023)) = o8;
    }
  }
  __syncthreads();

  const int wave = tid >> 5;
  const int lane = tid & 31;
  const int hi = lane >> 4;
  const int c16 = lane & 15;
  int colv[NPAN];
  float bc[NPAN];
#pragma unroll
  for (int j = 0; j < NPAN; ++j) {
    colv[j] = n0 + j * 16 + c16;
    bc[j] = bias[min(colv[j], VOCAB - 1)];
  }

  // 128 M tiles as 64 pairs; each wave owns pairs p = wave, wave+8, ...
  for (int p = wave; p < (BB * TSEQ) / 32; p += 8) {
    const int m0 = p * 32;
    if (p + 8 < (BB * TSEQ) / 32)  // hint next A panel toward the caches
      __builtin_prefetch(A + (size_t)(m0 + 256) * CDIM + lane * 128, 0, 1);

    const bf16* arow0 = A + (size_t)m0 * CDIM;
    const bf16* arow1 = arow0 + 16 * CDIM;
    v8f acc0[NPAN], acc1[NPAN];
#pragma unroll
    for (int j = 0; j < NPAN; ++j) { acc0[j] = (v8f)0.f; acc1[j] = (v8f)0.f; }

#pragma unroll 2
    for (int kk = 0; kk < CDIM; kk += 32) {
      v16bf a0 = load_a_frag(arow0 + kk, CDIM, lane);
      v16bf a1 = load_a_frag(arow1 + kk, CDIM, lane);
#pragma unroll
      for (int j = 0; j < NPAN; ++j) {
        v16bf bj = lds_b_frag_sw(bS + j * 16384, kk, lane);
        acc0[j] = wmma_bf16(a0, bj, acc0[j]);
        acc1[j] = wmma_bf16(a1, bj, acc1[j]);
      }
    }

#pragma unroll
    for (int g = 0; g < 8; ++g) {
      const size_t r0 = (size_t)(m0 + hi * 8 + g) * VOCAB;
      const size_t r1 = (size_t)(m0 + 16 + hi * 8 + g) * VOCAB;
#pragma unroll
      for (int j = 0; j < NPAN; ++j) {
        if (colv[j] < VOCAB) {
          out[r0 + colv[j]] = acc0[j][g] + bc[j];
          out[r1 + colv[j]] = acc1[j][g] + bc[j];
        }
      }
    }
  }
}

// ---------------------------------------------------------------------------
extern "C" void kernel_launch(void* const* d_in, const int* in_sizes, int n_in,
                              void* d_out, int out_size, void* d_ws, size_t ws_size,
                              hipStream_t stream) {
  const int*   idx     = (const int*)d_in[0];
  const float* tok_emb = (const float*)d_in[1];
  const float* pos_emb = (const float*)d_in[2];
  const float* Wq      = (const float*)d_in[3];
  const float* Wk      = (const float*)d_in[4];
  const float* Wv      = (const float*)d_in[5];
  const float* W_lm    = (const float*)d_in[6];
  const float* b_lm    = (const float*)d_in[7];
  float* out = (float*)d_out;

  char* ws = (char*)d_ws;
  const size_t SZ_X  = (size_t)BB * TSEQ * CDIM * sizeof(bf16);     // 4 MB
  const size_t SZ_W  = (size_t)NH * HD * CDIM * sizeof(bf16);       // 2 MB
  const size_t SZ_QK = (size_t)BB * NH * TSEQ * HD * sizeof(bf16);  // 4 MB
  bf16* xbf = (bf16*)(ws);
  bf16* wqT = (bf16*)(ws + SZ_X);
  bf16* wkT = (bf16*)(ws + SZ_X + SZ_W);
  bf16* wvT = (bf16*)(ws + SZ_X + 2 * SZ_W);
  bf16* qb  = (bf16*)(ws + SZ_X + 3 * SZ_W);
  bf16* kb  = (bf16*)(ws + SZ_X + 3 * SZ_W + SZ_QK);
  bf16* vTb = (bf16*)(ws + SZ_X + 3 * SZ_W + 2 * SZ_QK);
  bf16* ao  = (bf16*)(ws + SZ_X + 3 * SZ_W + 3 * SZ_QK);

  k_embed<<<BB * TSEQ, 256, 0, stream>>>(idx, tok_emb, pos_emb, xbf);
  k_convw<<<(NH * CDIM * HD) / 256, 256, 0, stream>>>(Wq, Wk, Wv, wqT, wkT, wvT);
  k_qkv<<<dim3(TSEQ / 16, NH, BB), 128, 0, stream>>>(xbf, wqT, wkT, wvT, qb, kb, vTb);
  k_attn<<<dim3(TSEQ / 16, NH, BB), 32, 0, stream>>>(qb, kb, vTb, ao);
  k_lmhead<<<(VOCAB + NPAN * 16 - 1) / (NPAN * 16), 256, 0, stream>>>(ao, W_lm, b_lm, out);
}
